// ClassifierKernels_70042326663878
// MI455X (gfx1250) — matrix operand order repaired and checked by reading, not compiled, not run on hardware
//
#include <hip/hip_runtime.h>
#include <hip/hip_bf16.h>
#include <cstdint>
#include <cstddef>

typedef _Float16 v8h  __attribute__((ext_vector_type(8)));
typedef _Float16 v16h __attribute__((ext_vector_type(16)));
typedef float    v8f  __attribute__((ext_vector_type(8)));

#define BS   8
#define H    512
#define W    512
#define H1   510      // after conv1 (3x3 VALID)
#define W1   510
#define HO   508      // after conv2 (3x3 VALID)
#define WO   508
#define CMID 32
#define NK   8

// workspace layout
#define GRAY_BYTES ((size_t)BS * H * W * 4)          // 8 MB f32
#define W2F_OFF    (GRAY_BYTES)                      // 18432 B f16 swizzled conv2 weights
#define Z1_OFF     (GRAY_BYTES + 32768)              // f16 channels-last z1: 8*510*510*32

// ---------------------------------------------------------------------------
// gray = sum over 3 mosaic channels
// ---------------------------------------------------------------------------
__global__ __launch_bounds__(256) void gray_kernel(const float* __restrict__ mosaic,
                                                   float* __restrict__ gray) {
    int tid = blockIdx.x * 256 + threadIdx.x;
    if (tid >= BS * H * W) return;
    int b = tid >> 18;                 // H*W = 262144
    int pix = tid & (H * W - 1);
    const float* mb = mosaic + (size_t)b * 3 * H * W;
    gray[tid] = mb[pix] + mb[H * W + pix] + mb[2 * H * W + pix];
}

// ---------------------------------------------------------------------------
// Swizzle conv2 weights (32,32,3,3) f32 -> per-lane WMMA B fragments f16:
// layout [tap(9)][Nhalf(2)][lane(32)][h(16)], lane = (Kgroup<<4)|Ncol,
// K = h<8 ? Kgroup*8 + h : 16 + Kgroup*8 + (h-8)  (B 32x16: lanes 0-15 hold K 0..7|16..23, lanes 16-31 K 8..15|24..31)
// ---------------------------------------------------------------------------
__global__ __launch_bounds__(256) void w2frag_kernel(const float* __restrict__ w2,
                                                     _Float16* __restrict__ w2f) {
    int tid = blockIdx.x * 256 + threadIdx.x;
    if (tid >= 9 * 2 * 32 * 16) return;
    int h    = tid & 15;
    int lane = (tid >> 4) & 31;
    int nh   = (tid >> 9) & 1;
    int t    = tid >> 10;
    int g    = lane >> 4;
    int n    = nh * 16 + (lane & 15);   // output channel (N column)
    int k    = (h < 8) ? (g * 8 + h) : (16 + g * 8 + (h - 8));   // input channel (K row)
    int dy = t / 3, dx = t % 3;
    w2f[tid] = (_Float16)w2[((n * 32 + k) * 3 + dy) * 3 + dx];
}

// ---------------------------------------------------------------------------
// conv1 (5ch incl. xg/yg grids -> 32ch, 3x3 VALID) + ReLU, output f16
// channels-last: z1[((b*510+y)*510+x)*32 + c]
// ---------------------------------------------------------------------------
__global__ __launch_bounds__(256) void conv1_kernel(const float* __restrict__ mosaic,
                                                    const float* __restrict__ w1,
                                                    const float* __restrict__ b1,
                                                    _Float16* __restrict__ z1) {
    __shared__ float sW1[32 * 45];
    __shared__ float sB1[32];
    for (int i = threadIdx.x; i < 32 * 45; i += 256) sW1[i] = w1[i];
    if (threadIdx.x < 32) sB1[threadIdx.x] = b1[threadIdx.x];
    __syncthreads();

    int tid = blockIdx.x * 256 + threadIdx.x;
    if (tid >= BS * H1 * W1) return;
    int b = tid / (H1 * W1);
    int rem = tid % (H1 * W1);
    int y = rem / W1, x = rem % W1;

    const float* mb = mosaic + (size_t)b * 3 * H * W;
    float inv[45];
#pragma unroll
    for (int ky = 0; ky < 3; ky++)
#pragma unroll
        for (int kx = 0; kx < 3; kx++) {
            int off = (y + ky) * W + (x + kx);
            inv[0 * 9 + ky * 3 + kx] = mb[off];
            inv[1 * 9 + ky * 3 + kx] = mb[H * W + off];
            inv[2 * 9 + ky * 3 + kx] = mb[2 * H * W + off];
            inv[3 * 9 + ky * 3 + kx] = (float)((x + kx) & 1);  // xg
            inv[4 * 9 + ky * 3 + kx] = (float)((y + ky) & 1);  // yg
        }

    _Float16 hbuf[32];
    for (int c = 0; c < 32; c++) {
        float acc = sB1[c];
        const float* wc = &sW1[c * 45];
#pragma unroll
        for (int i = 0; i < 45; i++) acc = fmaf(inv[i], wc[i], acc);
        hbuf[c] = (_Float16)fmaxf(acc, 0.f);
    }
    _Float16* zp = z1 + ((size_t)tid << 5);
#pragma unroll
    for (int i = 0; i < 4; i++) *(v8h*)(zp + i * 8) = *(const v8h*)(hbuf + i * 8);
}

// ---------------------------------------------------------------------------
// Fused: conv2 (implicit GEMM via WMMA f16) + bias + ReLU + conv3 + softmax +
// tile/kernel einsum + Bayer compose. One wave32 per (b, y, 16-pixel x tile).
// ---------------------------------------------------------------------------
__global__ __launch_bounds__(256) void main_kernel(const _Float16* __restrict__ z1,
                                                   const _Float16* __restrict__ w2f,
                                                   const float* __restrict__ b2,
                                                   const float* __restrict__ w3,
                                                   const float* __restrict__ b3,
                                                   const float* __restrict__ kern,
                                                   const float* __restrict__ gray,
                                                   float* __restrict__ out) {
    __shared__ _Float16 sW2[9][2][32][16];   // 18432 B: swizzled B fragments
    __shared__ float    sB2[32];
    __shared__ float    sW3[8][32];
    __shared__ float    sB3[8];
    __shared__ float    sKern[2][8][25];
    __shared__ float    sZ2[8][16][32];      // per-wave conv2 output staging

    // cooperative LDS fill
    {
        const uint32_t* src = (const uint32_t*)w2f;
        uint32_t* dst = (uint32_t*)&sW2[0][0][0][0];
        for (int i = threadIdx.x; i < 9 * 2 * 32 * 16 / 2; i += 256) dst[i] = src[i];
        if (threadIdx.x < 32) sB2[threadIdx.x] = b2[threadIdx.x];
        for (int i = threadIdx.x; i < 256; i += 256) ((float*)sW3)[i] = w3[i];
        if (threadIdx.x < 8) sB3[threadIdx.x] = b3[threadIdx.x];
        for (int i = threadIdx.x; i < 400; i += 256) ((float*)sKern)[i] = kern[i];
    }
    __syncthreads();

    const int lane = threadIdx.x & 31;
    const int wave = threadIdx.x >> 5;
    const int g = lane >> 4;        // A-layout K-group / D-layout row group
    const int r = lane & 15;        // A-layout pixel row / D-layout N column

    const int tileId = blockIdx.x * 8 + wave;   // exactly BS*HO*32 tiles
    const int b  = tileId / (HO * 32);
    const int rem = tileId % (HO * 32);
    const int y  = rem >> 5;
    const int x0 = (rem & 31) << 4;

    v8f acc0 = {};   // N = 0..15
    v8f acc1 = {};   // N = 16..31

#pragma unroll
    for (int t = 0; t < 9; t++) {
        const int dy = t / 3, dx = t % 3;
        int xx = x0 + r + dx;
        if (xx > W1 - 1) xx = W1 - 1;            // clamp (tail pixels masked at store)
        const _Float16* ap = z1 + ((size_t)((b * H1 + (y + dy)) * W1 + xx) << 5);
        // A fragment: lanes 0-15: K 0..7 | 16..23 ; lanes 16-31: K 8..15 | 24..31
        v8h alo = *(const v8h*)(ap + g * 8);
        v8h ahi = *(const v8h*)(ap + 16 + g * 8);
        const _Float16* bp0 = &sW2[t][0][lane][0];
        const _Float16* bp1 = &sW2[t][1][lane][0];
        v8h b0l = *(const v8h*)bp0, b0h = *(const v8h*)(bp0 + 8);
        v8h b1l = *(const v8h*)bp1, b1h = *(const v8h*)(bp1 + 8);
        v16h A, B0, B1;
#pragma unroll
        for (int i = 0; i < 8; i++) {
            A[i]  = alo[i]; A[i + 8]  = ahi[i];
            B0[i] = b0l[i]; B0[i + 8] = b0h[i];
            B1[i] = b1l[i]; B1[i + 8] = b1h[i];
        }
        acc0 = __builtin_amdgcn_wmma_f32_16x16x32_f16(false, A, false, B0, (short)0, acc0, false, false);
        acc1 = __builtin_amdgcn_wmma_f32_16x16x32_f16(false, A, false, B1, (short)0, acc1, false, false);
    }

    // bias + ReLU, scatter D (lane=N col, VGPR j = row M or M+8) to LDS
    {
        const int mb_ = g ? 8 : 0;
        const float bb0 = sB2[r];
        const float bb1 = sB2[16 + r];
#pragma unroll
        for (int j = 0; j < 8; j++) {
            sZ2[wave][mb_ + j][r]      = fmaxf(acc0[j] + bb0, 0.f);
            sZ2[wave][mb_ + j][16 + r] = fmaxf(acc1[j] + bb1, 0.f);
        }
    }
    __syncthreads();

    // tail: lanes 0-15 each own one pixel of the tile
    if (lane < 16) {
        const int x = x0 + lane;
        if (x < WO) {
            float z[32];
#pragma unroll
            for (int c = 0; c < 32; c++) z[c] = sZ2[wave][lane][c];

            // conv3 1x1 -> 8 logits, softmax
            float wgt[8];
            {
                float lg[8];
#pragma unroll
                for (int o = 0; o < 8; o++) {
                    float s = sB3[o];
#pragma unroll
                    for (int c = 0; c < 32; c++) s = fmaf(sW3[o][c], z[c], s);
                    lg[o] = s;
                }
                float mx = lg[0];
#pragma unroll
                for (int o = 1; o < 8; o++) mx = fmaxf(mx, lg[o]);
                float sum = 0.f;
#pragma unroll
                for (int o = 0; o < 8; o++) { wgt[o] = __expf(lg[o] - mx); sum += wgt[o]; }
                const float invs = 1.f / sum;
#pragma unroll
                for (int o = 0; o < 8; o++) wgt[o] *= invs;
            }

            // einsum: r_c = sum_s gray_tile[s] * (sum_n kern[c][n][s] * wgt[n])
            const float* gb = gray + (size_t)b * H * W;
            float r0 = 0.f, r1 = 0.f;
#pragma unroll
            for (int di = 0; di < 5; di++) {
#pragma unroll
                for (int dj = 0; dj < 5; dj++) {
                    const int s = di * 5 + dj;
                    const float gv = gb[(y + di) * W + (x + dj)];
                    float k0 = 0.f, k1 = 0.f;
#pragma unroll
                    for (int n = 0; n < 8; n++) {
                        k0 = fmaf(sKern[0][n][s], wgt[n], k0);
                        k1 = fmaf(sKern[1][n][s], wgt[n], k1);
                    }
                    r0 = fmaf(gv, k0, r0);
                    r1 = fmaf(gv, k1, r1);
                }
            }

            // Bayer compose: crop offset is (2,2) -> parities are x&1, y&1
            const float gc = gb[(y + 2) * W + (x + 2)];
            const int jx = x & 1, iy = y & 1;
            const float green = (jx == iy) ? gc : (jx ? r0 : r1);

            out[(((size_t)(b * 3 + 0) * HO + y) * WO) + x] = 0.f;
            out[(((size_t)(b * 3 + 1) * HO + y) * WO) + x] = green;
            out[(((size_t)(b * 3 + 2) * HO + y) * WO) + x] = 0.f;
        }
    }
}

// ---------------------------------------------------------------------------
extern "C" void kernel_launch(void* const* d_in, const int* in_sizes, int n_in,
                              void* d_out, int out_size, void* d_ws, size_t ws_size,
                              hipStream_t stream) {
    const float* mosaic = (const float*)d_in[0];
    // d_in[1] = mask (unused by the reference math)
    const float* w1   = (const float*)d_in[2];
    const float* b1   = (const float*)d_in[3];
    const float* w2   = (const float*)d_in[4];
    const float* b2   = (const float*)d_in[5];
    const float* w3   = (const float*)d_in[6];
    const float* b3   = (const float*)d_in[7];
    const float* kern = (const float*)d_in[8];
    float* out = (float*)d_out;

    char* ws = (char*)d_ws;
    float*    gray = (float*)ws;
    _Float16* w2f  = (_Float16*)(ws + W2F_OFF);
    _Float16* z1   = (_Float16*)(ws + Z1_OFF);

    gray_kernel<<<(BS * H * W + 255) / 256, 256, 0, stream>>>(mosaic, gray);
    w2frag_kernel<<<(9 * 2 * 32 * 16 + 255) / 256, 256, 0, stream>>>(w2, w2f);
    conv1_kernel<<<(BS * H1 * W1 + 255) / 256, 256, 0, stream>>>(mosaic, w1, b1, z1);
    main_kernel<<<(BS * HO * 32) / 8, 256, 0, stream>>>(z1, w2f, b2, w3, b3, kern, gray, out);
}
